// MaskedAttentionBlock_31190052504016
// MI455X (gfx1250) — matrix operand order repaired and checked by reading, not compiled, not run on hardware
//
#include <hip/hip_runtime.h>

// ---------------------------------------------------------------------------
// MI455X / gfx1250 masked-attention transformer block.
//   * all matmuls on v_wmma_f32_16x16x32_bf16 (fp32 accum)
//   * GEMM tiles staged by the Tensor Data Mover (tensor_load_to_lds) with
//     hardware LDS row padding, double-buffered, synced via s_wait_tensorcnt
//   * flash-style online softmax: never materializes [B,h,S,S]
//   * weights pre-transposed to [N][K] bf16; V written transposed [b][h][d][s]
//   * adjacency mask bit-packed (32 keys/word)
// ---------------------------------------------------------------------------

typedef __bf16 bf16;
typedef __attribute__((ext_vector_type(16))) __bf16 v16bf;
typedef __attribute__((ext_vector_type(8)))  float  v8f;
typedef __attribute__((ext_vector_type(4)))  unsigned v4u;
typedef __attribute__((ext_vector_type(8)))  int      v8i;
typedef __attribute__((ext_vector_type(4)))  int      v4i;

#define WMMA_BF16(Af, Bf, Cf) \
  __builtin_amdgcn_wmma_f32_16x16x32_bf16(false, (Af), false, (Bf), (short)0, (Cf), false, false)

#if __has_builtin(__builtin_amdgcn_tensor_load_to_lds) && \
    __has_builtin(__builtin_amdgcn_s_wait_tensorcnt)
#define USE_TDM 1
#else
#define USE_TDM 0
#endif

static constexpr int kB     = 2;
static constexpr int kS     = 2048;
static constexpr int kH     = 512;
static constexpr int kHeads = 8;
static constexpr int kHD    = 64;
static constexpr int kDFF   = 2048;
static constexpr int kT     = kB * kS;   // 4096 tokens
static constexpr float kNeg = -1e9f;

__device__ __forceinline__ bf16 f2bf(float f) {
  unsigned u = __builtin_bit_cast(unsigned, f);
  u = (u + 0x7FFFu + ((u >> 16) & 1u)) >> 16;   // round-to-nearest-even
  unsigned short s = (unsigned short)u;
  return __builtin_bit_cast(bf16, s);
}

union Frag {              // one WMMA 16-bit operand: 16 bf16 = 8 VGPRs
  v16bf v;
  uint4 q[2];
};

#if USE_TDM
// Issue one TDM 2-D tile load: rows x 32 bf16, row stride = strideElems,
// LDS destination padded 16B after every 64B row (-> [rows][40] bf16 layout).
// D# layout per CDNA5 ISA 8.3/8.4 (group2/3 zero => 2-D tensor).
// This toolchain's builtin takes 6 args (g0, g1, g2, g3, g-extra, cpol).
__device__ __forceinline__ void tdm_load_tile(unsigned ldsAddr,
                                              const bf16* gsrc,
                                              unsigned tdim0, unsigned tdim1,
                                              unsigned tileRows,
                                              unsigned strideElems) {
  unsigned long long ga = (unsigned long long)(uintptr_t)gsrc;
  v4u g0;
  g0[0] = 1u;                                   // count=1, user descriptor
  g0[1] = ldsAddr;                              // LDS byte address
  g0[2] = (unsigned)ga;                         // global_addr[31:0]
  g0[3] = (unsigned)(ga >> 32) | (2u << 30);    // global_addr[56:32] | type=2
  v8i g1;
  const unsigned w0 = (1u << 16)   // data_size = 2 bytes
                    | (1u << 20)   // pad_enable
                    | (3u << 22)   // pad_interval = 16 DWORDs (64B row)
                    | (3u << 25);  // pad_amount   = 4 DWORDs (16B)
  g1[0] = (int)w0;
  g1[1] = (int)((tdim0 & 0xFFFFu) << 16);                    // tensor_dim0 lo
  g1[2] = (int)((tdim0 >> 16) | ((tdim1 & 0xFFFFu) << 16));  // dim0 hi|dim1 lo
  g1[3] = (int)((tdim1 >> 16) | (32u << 16));                // dim1 hi|tile_dim0=32
  g1[4] = (int)tileRows;                                     // tile_dim1 (dim2=0)
  g1[5] = (int)strideElems;                                  // dim0_stride lo
  g1[6] = 0;
  g1[7] = 0;
  v4i z4 = {0, 0, 0, 0};
  v8i z8 = {0, 0, 0, 0, 0, 0, 0, 0};
  __builtin_amdgcn_tensor_load_to_lds(g0, g1, z4, z4, z8, 0);
}
#endif

// ---------------- fp32 -> bf16 transposed weight conversion ---------------
__global__ void cvt_transpose_kernel(const float* __restrict__ W,
                                     bf16* __restrict__ Wt, int K, int N) {
  __shared__ bf16 t[32][33];
  const int tid = threadIdx.x;
  const int tx = tid & 31, ty = tid >> 5;
  const int k0 = blockIdx.y * 32, n0 = blockIdx.x * 32;
#pragma unroll
  for (int r = ty; r < 32; r += 8)
    t[r][tx] = f2bf(W[(size_t)(k0 + r) * N + n0 + tx]);
  __syncthreads();
#pragma unroll
  for (int r = ty; r < 32; r += 8)
    Wt[(size_t)(n0 + r) * K + k0 + tx] = t[tx][r];
}

// ---------------- adjacency mask bit-pack (32 keys per uint32) ------------
__global__ void pack_mask_kernel(const unsigned char* __restrict__ adj,
                                 unsigned* __restrict__ out) {
  const int nw = kB * kS * (kS / 32);
  for (int wi = blockIdx.x * blockDim.x + threadIdx.x; wi < nw;
       wi += gridDim.x * blockDim.x) {
    const unsigned char* p = adj + (size_t)wi * 32;
    uint4 a = ((const uint4*)p)[0];
    uint4 b = ((const uint4*)p)[1];
    const unsigned char* ea = (const unsigned char*)&a;
    const unsigned char* eb = (const unsigned char*)&b;
    unsigned m = 0;
#pragma unroll
    for (int i = 0; i < 16; ++i) m |= (unsigned)(ea[i] != 0) << i;
#pragma unroll
    for (int i = 0; i < 16; ++i) m |= (unsigned)(eb[i] != 0) << (16 + i);
    out[wi] = m;
  }
}

// --------------------------- LayerNorm -> bf16 -----------------------------
__global__ void layernorm_bf16_kernel(const float* __restrict__ x,
                                      const float* __restrict__ g,
                                      const float* __restrict__ b,
                                      bf16* __restrict__ out) {
  const int tok = blockIdx.x;
  const int tid = threadIdx.x;
  const int lane = tid & 31, w = tid >> 5;
  const float* row = x + (size_t)tok * kH;
  float2 v = ((const float2*)row)[tid];
  float s = v.x + v.y, ss = v.x * v.x + v.y * v.y;
#pragma unroll
  for (int m = 16; m >= 1; m >>= 1) {
    s  += __shfl_xor(s,  m, 32);
    ss += __shfl_xor(ss, m, 32);
  }
  __shared__ float rs[8], rss[8], stat[2];
  if (lane == 0) { rs[w] = s; rss[w] = ss; }
  __syncthreads();
  if (tid == 0) {
    float S = 0.f, SS = 0.f;
#pragma unroll
    for (int i = 0; i < 8; ++i) { S += rs[i]; SS += rss[i]; }
    float mu = S * (1.0f / kH);
    float var = SS * (1.0f / kH) - mu * mu;
    stat[0] = mu;
    stat[1] = rsqrtf(var + 1e-5f);
  }
  __syncthreads();
  const float mu = stat[0], rinv = stat[1];
  const int c = 2 * tid;
  out[(size_t)tok * kH + c + 0] = f2bf((v.x - mu) * rinv * g[c + 0] + b[c + 0]);
  out[(size_t)tok * kH + c + 1] = f2bf((v.y - mu) * rinv * g[c + 1] + b[c + 1]);
}

// --------------------------- bf16 tiled GEMM -------------------------------
// C[M,N] = act(A[M,K] @ B[K,N] + bias (+res)); B given TRANSPOSED: Bt[N][K].
// 256 thr, 128x64 block tile, 8 waves as 4x2, wave = 32x32 = 4 WMMA / K-step.
// Tiles staged by TDM (double-buffered, TENSORcnt) when available.
// OUTMODE: 0 = fp32, 1 = bf16, 2 = bf16 scattered to V^T [b][h][d][s]
template <bool GELU, int OUTMODE, bool HAS_RES>
__global__ void gemm_bf16_kernel(const bf16* __restrict__ A,
                                 const bf16* __restrict__ Bt,
                                 const float* __restrict__ bias,
                                 const float* __restrict__ res,
                                 void* __restrict__ Cout,
                                 int M, int N, int K) {
#if USE_TDM
  __shared__ __align__(16) bf16 sA[2][128][40];
  __shared__ __align__(16) bf16 sB[2][64][40];
#else
  __shared__ __align__(16) bf16 sA[1][128][40];
  __shared__ __align__(16) bf16 sB[1][64][40];
#endif
  const int tid = threadIdx.x;
  const int lane = tid & 31, w = tid >> 5;
  const int hi = lane >> 4, ll = lane & 15;
  const int wr = w >> 1, wc = w & 1;
  const int m0 = blockIdx.y * 128, n0 = blockIdx.x * 64;

  v8f acc00 = {}, acc01 = {}, acc10 = {}, acc11 = {};

#if USE_TDM
  const unsigned ldsA0 = (unsigned)(uintptr_t)&sA[0][0][0];
  const unsigned ldsA1 = (unsigned)(uintptr_t)&sA[1][0][0];
  const unsigned ldsB0 = (unsigned)(uintptr_t)&sB[0][0][0];
  const unsigned ldsB1 = (unsigned)(uintptr_t)&sB[1][0][0];
  if (w == 0) {   // prologue: fill buffer 0 (k0 = 0)
    tdm_load_tile(ldsA0, A  + (size_t)m0 * K, K, M, 128, K);
    tdm_load_tile(ldsB0, Bt + (size_t)n0 * K, K, N, 64,  K);
  }
  int cur = 0;
  for (int k0 = 0; k0 < K; k0 += 32, cur ^= 1) {
    if (w == 0) {
      if (k0 + 32 < K) {   // kick off next slice into the other buffer
        tdm_load_tile(cur ? ldsA0 : ldsA1, A  + (size_t)m0 * K + k0 + 32, K, M, 128, K);
        tdm_load_tile(cur ? ldsB0 : ldsB1, Bt + (size_t)n0 * K + k0 + 32, K, N, 64,  K);
        __builtin_amdgcn_s_wait_tensorcnt(2);  // retire current 2 (in-order)
      } else {
        __builtin_amdgcn_s_wait_tensorcnt(0);
      }
    }
    __syncthreads();   // release all waves once wave0's TDM data landed
#else
  const int ar = tid >> 1, ac = (tid & 1) * 16;
  const int br = tid >> 2, bc = (tid & 3) * 8;
  const int cur = 0;
  for (int k0 = 0; k0 < K; k0 += 32) {
    *(uint4*)&sA[0][ar][ac] =
        *(const uint4*)(A + (size_t)(m0 + ar) * K + k0 + ac);
    *(uint4*)&sA[0][ar][ac + 8] =
        *(const uint4*)(A + (size_t)(m0 + ar) * K + k0 + ac + 8);
    *(uint4*)&sB[0][br][bc] =
        *(const uint4*)(Bt + (size_t)(n0 + br) * K + k0 + bc);
    __syncthreads();
#endif

    Frag a0, a1, b0, b1;
    a0.q[0] = *(const uint4*)&sA[cur][wr * 32 + ll][hi * 8];
    a0.q[1] = *(const uint4*)&sA[cur][wr * 32 + ll][16 + hi * 8];
    a1.q[0] = *(const uint4*)&sA[cur][wr * 32 + 16 + ll][hi * 8];
    a1.q[1] = *(const uint4*)&sA[cur][wr * 32 + 16 + ll][16 + hi * 8];
    b0.q[0] = *(const uint4*)&sB[cur][wc * 32 + ll][hi * 8];
    b0.q[1] = *(const uint4*)&sB[cur][wc * 32 + ll][16 + hi * 8];
    b1.q[0] = *(const uint4*)&sB[cur][wc * 32 + 16 + ll][hi * 8];
    b1.q[1] = *(const uint4*)&sB[cur][wc * 32 + 16 + ll][16 + hi * 8];

    acc00 = WMMA_BF16(a0.v, b0.v, acc00);
    acc01 = WMMA_BF16(a0.v, b1.v, acc01);
    acc10 = WMMA_BF16(a1.v, b0.v, acc10);
    acc11 = WMMA_BF16(a1.v, b1.v, acc11);
    __syncthreads();
  }

  bf16*  outb = (bf16*)Cout;
  float* outf = (float*)Cout;
#pragma unroll
  for (int i = 0; i < 2; ++i) {
#pragma unroll
    for (int j = 0; j < 2; ++j) {
      const v8f& acc = i ? (j ? acc11 : acc10) : (j ? acc01 : acc00);
#pragma unroll
      for (int r = 0; r < 8; ++r) {
        const int mrow = m0 + wr * 32 + i * 16 + r + 8 * hi;
        const int ncol = n0 + wc * 32 + j * 16 + ll;
        float v = acc[r] + bias[ncol];
        if (HAS_RES) v += res[(size_t)mrow * N + ncol];
        if (GELU)    v = 0.5f * v * (1.0f + erff(v * 0.70710678118654752f));
        if (OUTMODE == 0) {
          outf[(size_t)mrow * N + ncol] = v;
        } else if (OUTMODE == 1) {
          outb[(size_t)mrow * N + ncol] = f2bf(v);
        } else {  // scatter into V^T [b][h][d][s]
          const int bb = mrow >> 11, s = mrow & (kS - 1);
          const int hh = ncol >> 6,  d = ncol & (kHD - 1);
          outb[(((size_t)bb * kHeads + hh) * kHD + d) * kS + s] = f2bf(v);
        }
      }
    }
  }
}

// --------------------------- flash attention -------------------------------
__global__ void attention_kernel(const bf16* __restrict__ Q,
                                 const bf16* __restrict__ Kmat,
                                 const bf16* __restrict__ Vt,
                                 const unsigned* __restrict__ mpack,
                                 bf16* __restrict__ Oout) {
  const int b = blockIdx.z, h = blockIdx.y;
  const int tid = threadIdx.x;
  const int w = tid >> 5, lane = tid & 31, hi = lane >> 4, ll = lane & 15;
  const int q0 = blockIdx.x * 64 + w * 16;
  const float scale = 0.125f;   // 1/sqrt(64)

  __shared__ __align__(16) bf16 sP[4][16][40];

  auto tptr = [&](int tok) -> size_t {
    return ((size_t)b * kS + tok) * kH + h * kHD;
  };
  const bf16* vhead = Vt + ((size_t)(b * kHeads + h)) * kHD * kS;

  Frag qf0, qf1;
  {
    const bf16* qp = Q + tptr(q0 + ll);
    qf0.q[0] = *(const uint4*)(qp + hi * 16);
    qf0.q[1] = *(const uint4*)(qp + hi * 16 + 8);
    qf1.q[0] = *(const uint4*)(qp + 32 + hi * 16);
    qf1.q[1] = *(const uint4*)(qp + 40 + hi * 16);
  }

  v8f zero = {};
  v8f O0 = zero, O1 = zero, O2 = zero, O3 = zero;
  float mrow[8], lrow[8];
#pragma unroll
  for (int r = 0; r < 8; ++r) { mrow[r] = -1e30f; lrow[r] = 0.f; }

  for (int kt0 = 0; kt0 < kS; kt0 += 32) {
    Frag k00, k01, k10, k11, vf0, vf1, vf2, vf3;
    {
      const bf16* kp0 = Kmat + tptr(kt0 + ll) + hi * 16;
      const bf16* kp1 = Kmat + tptr(kt0 + 16 + ll) + hi * 16;
      k00.q[0] = ((const uint4*)kp0)[0];        k00.q[1] = ((const uint4*)kp0)[1];
      k01.q[0] = ((const uint4*)(kp0 + 32))[0]; k01.q[1] = ((const uint4*)(kp0 + 32))[1];
      k10.q[0] = ((const uint4*)kp1)[0];        k10.q[1] = ((const uint4*)kp1)[1];
      k11.q[0] = ((const uint4*)(kp1 + 32))[0]; k11.q[1] = ((const uint4*)(kp1 + 32))[1];
    }
    {
      const bf16* vp = vhead + (size_t)ll * kS + kt0 + hi * 16;
      vf0.q[0] = ((const uint4*)vp)[0];
      vf0.q[1] = ((const uint4*)(vp + 8))[0];
      vf1.q[0] = ((const uint4*)(vp + 16 * kS))[0];
      vf1.q[1] = ((const uint4*)(vp + 16 * kS + 8))[0];
      vf2.q[0] = ((const uint4*)(vp + 32 * kS))[0];
      vf2.q[1] = ((const uint4*)(vp + 32 * kS + 8))[0];
      vf3.q[0] = ((const uint4*)(vp + 48 * kS))[0];
      vf3.q[1] = ((const uint4*)(vp + 48 * kS + 8))[0];
    }
    unsigned mw[8];
    {
      const unsigned* mp =
          mpack + ((size_t)b * kS + q0 + 8 * hi) * (kS / 32) + (kt0 >> 5);
#pragma unroll
      for (int r = 0; r < 8; ++r) mw[r] = mp[(size_t)r * (kS / 32)];
    }

    v8f sa0 = zero, sa1 = zero;
    sa0 = WMMA_BF16(qf0.v, k00.v, sa0);
    sa0 = WMMA_BF16(qf1.v, k01.v, sa0);
    sa1 = WMMA_BF16(qf0.v, k10.v, sa1);
    sa1 = WMMA_BF16(qf1.v, k11.v, sa1);

    float corr[8];
#pragma unroll
    for (int r = 0; r < 8; ++r) {
      float s0 = ((mw[r] >> ll) & 1u)        ? sa0[r] * scale : kNeg;
      float s1 = ((mw[r] >> (16 + ll)) & 1u) ? sa1[r] * scale : kNeg;
      float rm = fmaxf(s0, s1);
#pragma unroll
      for (int m = 8; m >= 1; m >>= 1) rm = fmaxf(rm, __shfl_xor(rm, m, 16));
      const float mn = fmaxf(mrow[r], rm);
      const float c  = __expf(mrow[r] - mn);
      const float p0 = __expf(s0 - mn);
      const float p1 = __expf(s1 - mn);
      float rsum = p0 + p1;
#pragma unroll
      for (int m = 8; m >= 1; m >>= 1) rsum += __shfl_xor(rsum, m, 16);
      lrow[r] = lrow[r] * c + rsum;
      mrow[r] = mn;
      corr[r] = c;
      sP[w][r + 8 * hi][ll]      = f2bf(p0);
      sP[w][r + 8 * hi][16 + ll] = f2bf(p1);
    }
#pragma unroll
    for (int r = 0; r < 8; ++r) {
      O0[r] *= corr[r]; O1[r] *= corr[r]; O2[r] *= corr[r]; O3[r] *= corr[r];
    }

    Frag pf;
    pf.q[0] = *(const uint4*)&sP[w][ll][hi * 16];
    pf.q[1] = *(const uint4*)&sP[w][ll][hi * 16 + 8];
    O0 = WMMA_BF16(pf.v, vf0.v, O0);
    O1 = WMMA_BF16(pf.v, vf1.v, O1);
    O2 = WMMA_BF16(pf.v, vf2.v, O2);
    O3 = WMMA_BF16(pf.v, vf3.v, O3);
  }

#pragma unroll
  for (int r = 0; r < 8; ++r) {
    const float inv = 1.0f / lrow[r];
    const size_t o = tptr(q0 + r + 8 * hi);
    Oout[o + 0 * 16 + ll] = f2bf(O0[r] * inv);
    Oout[o + 1 * 16 + ll] = f2bf(O1[r] * inv);
    Oout[o + 2 * 16 + ll] = f2bf(O2[r] * inv);
    Oout[o + 3 * 16 + ll] = f2bf(O3[r] * inv);
  }
}

// --------------------------- launch ---------------------------------------
static inline size_t alignup(size_t x) { return (x + 255) & ~(size_t)255; }

extern "C" void kernel_launch(void* const* d_in, const int* in_sizes, int n_in,
                              void* d_out, int out_size, void* d_ws, size_t ws_size,
                              hipStream_t stream) {
  (void)in_sizes; (void)n_in; (void)out_size; (void)ws_size;
  const float* x     = (const float*)d_in[0];
  const unsigned char* adj = (const unsigned char*)d_in[1];  // jax bool = 1 byte
  const float* ln1_g = (const float*)d_in[2];
  const float* ln1_b = (const float*)d_in[3];
  const float* ln2_g = (const float*)d_in[4];
  const float* ln2_b = (const float*)d_in[5];
  const float* Wq = (const float*)d_in[6];  const float* bq = (const float*)d_in[7];
  const float* Wk = (const float*)d_in[8];  const float* bk = (const float*)d_in[9];
  const float* Wv = (const float*)d_in[10]; const float* bv = (const float*)d_in[11];
  const float* Wo = (const float*)d_in[12]; const float* bo = (const float*)d_in[13];
  const float* W1 = (const float*)d_in[14]; const float* b1 = (const float*)d_in[15];
  const float* W2 = (const float*)d_in[16]; const float* b2 = (const float*)d_in[17];

  char* p = (char*)d_ws;
  auto carve = [&](size_t bytes) { char* r = p; p += alignup(bytes); return r; };
  bf16* nx1 = (bf16*)carve((size_t)kT * kH * 2);
  bf16* WqT = (bf16*)carve((size_t)kH * kH * 2);
  bf16* WkT = (bf16*)carve((size_t)kH * kH * 2);
  bf16* WvT = (bf16*)carve((size_t)kH * kH * 2);
  bf16* WoT = (bf16*)carve((size_t)kH * kH * 2);
  bf16* W1T = (bf16*)carve((size_t)kH * kDFF * 2);
  bf16* W2T = (bf16*)carve((size_t)kDFF * kH * 2);
  bf16* Qb  = (bf16*)carve((size_t)kT * kH * 2);
  bf16* Kb  = (bf16*)carve((size_t)kT * kH * 2);
  bf16* Vt  = (bf16*)carve((size_t)kT * kH * 2);       // [b][h][d][s]
  bf16* Ab  = (bf16*)carve((size_t)kT * kH * 2);
  bf16* nx2 = (bf16*)carve((size_t)kT * kH * 2);
  bf16* h1  = (bf16*)carve((size_t)kT * kDFF * 2);
  float*    x1 = (float*)   carve((size_t)kT * kH * 4);
  unsigned* mp = (unsigned*)carve((size_t)kB * kS * (kS / 32) * 4);

  cvt_transpose_kernel<<<dim3(kH / 32, kH / 32),  256, 0, stream>>>(Wq, WqT, kH, kH);
  cvt_transpose_kernel<<<dim3(kH / 32, kH / 32),  256, 0, stream>>>(Wk, WkT, kH, kH);
  cvt_transpose_kernel<<<dim3(kH / 32, kH / 32),  256, 0, stream>>>(Wv, WvT, kH, kH);
  cvt_transpose_kernel<<<dim3(kH / 32, kH / 32),  256, 0, stream>>>(Wo, WoT, kH, kH);
  cvt_transpose_kernel<<<dim3(kDFF / 32, kH / 32), 256, 0, stream>>>(W1, W1T, kH, kDFF);
  cvt_transpose_kernel<<<dim3(kH / 32, kDFF / 32), 256, 0, stream>>>(W2, W2T, kDFF, kH);

  pack_mask_kernel<<<1024, 256, 0, stream>>>(adj, mp);

  layernorm_bf16_kernel<<<kT, 256, 0, stream>>>(x, ln1_g, ln1_b, nx1);

  dim3 gproj(kH / 64, kT / 128);
  gemm_bf16_kernel<false, 1, false><<<gproj, 256, 0, stream>>>(nx1, WqT, bq, nullptr, Qb, kT, kH, kH);
  gemm_bf16_kernel<false, 1, false><<<gproj, 256, 0, stream>>>(nx1, WkT, bk, nullptr, Kb, kT, kH, kH);
  gemm_bf16_kernel<false, 2, false><<<gproj, 256, 0, stream>>>(nx1, WvT, bv, nullptr, Vt, kT, kH, kH);

  attention_kernel<<<dim3(kS / 64, kHeads, kB), 128, 0, stream>>>(Qb, Kb, Vt, mp, Ab);

  gemm_bf16_kernel<false, 0, true><<<gproj, 256, 0, stream>>>(Ab, WoT, bo, x, x1, kT, kH, kH);

  layernorm_bf16_kernel<<<kT, 256, 0, stream>>>(x1, ln2_g, ln2_b, nx2);

  gemm_bf16_kernel<true, 1, false><<<dim3(kDFF / 64, kT / 128), 256, 0, stream>>>(nx2, W1T, b1, nullptr, h1, kT, kDFF, kH);
  gemm_bf16_kernel<false, 0, true><<<gproj, 256, 0, stream>>>(h1, W2T, b2, x1, (float*)d_out, kT, kH, kDFF);
}
